// KimiLinearAttention_32066225832179
// MI455X (gfx1250) — compile-verified, matrix operand order checked by reading, not run on hardware
//
#include <hip/hip_runtime.h>
#include <hip/hip_bf16.h>

// ---------------------------------------------------------------------------
// Kimi linear attention for MI455X (gfx1250, wave32, WMMA).
// Fixed shapes: B=4, T=4096, D=2048, H=16, CS=64, DK=128.
// Pipeline (all on `stream`):
//   1) fp32 -> bf16 conversion of x and the 5 weight matrices
//   2) bf16 WMMA GEMMs (register double-buffered): q,k,v,g = x @ W^T + b
//   3) in-place RoPE+ELU+1 on q,k ; in-place sigmoid on g
//   4) chunked linear-attention scan (state S kept in WMMA accumulators)
//   5) final bf16 WMMA GEMM: out = attn @ Wo^T + bo       (fp32 out)
// Workspace: ~424 MB of d_ws.
// ---------------------------------------------------------------------------

#define BB 4
#define TT 4096
#define DD 2048
#define HH 16
#define CSZ 64
#define DKK 128
#define MM (BB * TT)   // 16384 rows

typedef __attribute__((ext_vector_type(16))) __bf16 v16bf;
typedef __attribute__((ext_vector_type(8)))  float  v8f;

union Frag { uint4 u4[2]; v16bf v; };

// ---------------------------------------------------------------------------
// fp32 -> bf16 elementwise conversion
// ---------------------------------------------------------------------------
__global__ __launch_bounds__(256) void f32_to_bf16_kernel(
    const float* __restrict__ in, __bf16* __restrict__ out, int n) {
  int i = blockIdx.x * 256 + threadIdx.x;
  if (i < n) out[i] = (__bf16)in[i];
}

// ---------------------------------------------------------------------------
// GEMM: C[M,N] = A[M,K] @ Bm[N,K]^T + bias[N]
// A, Bm bf16 row-major; fp32 accumulate; output bf16 or fp32.
// BM=BN=128, BK=32, 256 threads = 8 waves; wave = 4(M) x 2(N) WMMA tiles.
// Global loads for K-step k+1 are prefetched into registers while the WMMAs
// for K-step k run, so HBM latency overlaps the matrix pipe.
// ---------------------------------------------------------------------------
template <bool OUT_BF16>
__global__ __launch_bounds__(256) void gemm_bf16_wmma(
    const __bf16* __restrict__ A, const __bf16* __restrict__ Bm,
    const float* __restrict__ bias, void* __restrict__ Cout,
    int Mdim, int Ndim, int Kdim) {
  constexpr int BM = 128, BN = 128, BK = 32;
  __shared__ __align__(16) __bf16 sA[BM][BK];
  __shared__ __align__(16) __bf16 sB[BN][BK];

  const int tid  = threadIdx.x;
  const int lane = tid & 31;
  const int w    = tid >> 5;   // 0..7
  const int wm   = w & 1;      // 64-row half
  const int wn   = w >> 1;     // 32-col quarter
  const int bm   = blockIdx.y * BM;
  const int bn   = blockIdx.x * BN;
  const int half = lane >> 4;
  const int l16  = lane & 15;

  // staging geometry: 512 16B chunks per 128x32 tile; 2 per thread
  const int row0s = (tid + 0)   >> 2;
  const int row1s = (tid + 256) >> 2;
  const int c80   = ((tid + 0)   & 3) * 8;
  const int c81   = ((tid + 256) & 3) * 8;

  v8f acc[4][2];
#pragma unroll
  for (int i = 0; i < 4; ++i)
#pragma unroll
    for (int j = 0; j < 2; ++j) acc[i][j] = {};

  // prologue: fetch K-step 0 to registers, park in LDS
  uint4 ra0 = *reinterpret_cast<const uint4*>(A  + (size_t)(bm + row0s) * Kdim + c80);
  uint4 ra1 = *reinterpret_cast<const uint4*>(A  + (size_t)(bm + row1s) * Kdim + c81);
  uint4 rb0 = *reinterpret_cast<const uint4*>(Bm + (size_t)(bn + row0s) * Kdim + c80);
  uint4 rb1 = *reinterpret_cast<const uint4*>(Bm + (size_t)(bn + row1s) * Kdim + c81);
  *reinterpret_cast<uint4*>(&sA[row0s][c80]) = ra0;
  *reinterpret_cast<uint4*>(&sA[row1s][c81]) = ra1;
  *reinterpret_cast<uint4*>(&sB[row0s][c80]) = rb0;
  *reinterpret_cast<uint4*>(&sB[row1s][c81]) = rb1;
  __syncthreads();

  for (int k0 = 0; k0 < Kdim; k0 += BK) {
    const bool has_next = (k0 + BK) < Kdim;
    // fetch next K-slab into registers (overlaps the WMMAs below)
    if (has_next) {
      const int kn = k0 + BK;
      ra0 = *reinterpret_cast<const uint4*>(A  + (size_t)(bm + row0s) * Kdim + kn + c80);
      ra1 = *reinterpret_cast<const uint4*>(A  + (size_t)(bm + row1s) * Kdim + kn + c81);
      rb0 = *reinterpret_cast<const uint4*>(Bm + (size_t)(bn + row0s) * Kdim + kn + c80);
      rb1 = *reinterpret_cast<const uint4*>(Bm + (size_t)(bn + row1s) * Kdim + kn + c81);
      // and nudge the slab after that toward L2 (global_prefetch_b8)
      if (k0 + 2 * BK < Kdim) {
        __builtin_prefetch(A  + (size_t)(bm + row0s) * Kdim + k0 + 2 * BK + c80, 0, 1);
        __builtin_prefetch(Bm + (size_t)(bn + row0s) * Kdim + k0 + 2 * BK + c80, 0, 1);
      }
    }

    // batched fragment loads, then the 8-WMMA chain
    Frag af[4], bfv[2];
#pragma unroll
    for (int mt = 0; mt < 4; ++mt) {
      const __bf16* p = &sA[wm * 64 + mt * 16 + l16][half * 8];
      af[mt].u4[0] = *reinterpret_cast<const uint4*>(p);
      af[mt].u4[1] = *reinterpret_cast<const uint4*>(p + 16);
    }
#pragma unroll
    for (int nt = 0; nt < 2; ++nt) {
      const __bf16* p = &sB[wn * 32 + nt * 16 + l16][half * 8];
      bfv[nt].u4[0] = *reinterpret_cast<const uint4*>(p);
      bfv[nt].u4[1] = *reinterpret_cast<const uint4*>(p + 16);
    }
#pragma unroll
    for (int mt = 0; mt < 4; ++mt)
#pragma unroll
      for (int nt = 0; nt < 2; ++nt)
        acc[mt][nt] = __builtin_amdgcn_wmma_f32_16x16x32_bf16(
            false, af[mt].v, false, bfv[nt].v, (short)0, acc[mt][nt], false, false);

    __syncthreads();   // everyone done reading current LDS tile
    if (has_next) {
      *reinterpret_cast<uint4*>(&sA[row0s][c80]) = ra0;
      *reinterpret_cast<uint4*>(&sA[row1s][c81]) = ra1;
      *reinterpret_cast<uint4*>(&sB[row0s][c80]) = rb0;
      *reinterpret_cast<uint4*>(&sB[row1s][c81]) = rb1;
      __syncthreads(); // tile visible before next iteration reads it
    }
  }

  // epilogue: add bias, store
#pragma unroll
  for (int mt = 0; mt < 4; ++mt) {
#pragma unroll
    for (int nt = 0; nt < 2; ++nt) {
      int n = bn + wn * 32 + nt * 16 + l16;
      float bv = bias[n];
#pragma unroll
      for (int r = 0; r < 8; ++r) {
        int m = bm + wm * 64 + mt * 16 + r + half * 8;
        float val = acc[mt][nt][r] + bv;
        if (OUT_BF16)
          ((__bf16*)Cout)[(size_t)m * Ndim + n] = (__bf16)val;
        else
          ((float*)Cout)[(size_t)m * Ndim + n] = val;
      }
    }
  }
}

// ---------------------------------------------------------------------------
// In-place RoPE + ELU + 1 on a bf16 (B,T,D) buffer, per-head pairs.
// ---------------------------------------------------------------------------
__global__ __launch_bounds__(256) void rope_elu_inplace(__bf16* __restrict__ q,
                                                        int npairs) {
  int p = blockIdx.x * 256 + threadIdx.x;
  if (p >= npairs) return;
  int j    = p & 63;           // pair index within head (DK/2 = 64)
  int rest = p >> 6;
  int h    = rest & (HH - 1);
  int bt   = rest >> 4;        // b*T + t
  int t    = bt & (TT - 1);
  size_t base = (size_t)bt * DD + (size_t)h * DKK + 2 * j;
  float x1 = (float)q[base];
  float x2 = (float)q[base + 1];
  float inv_freq = __powf(10000.f, -(float)j / 64.f);
  float ang = (float)t * inv_freq;
  float s, c;
  __sincosf(ang, &s, &c);
  float r1 = x1 * c - x2 * s;
  float r2 = x1 * s + x2 * c;
  q[base]     = (__bf16)(r1 > 0.f ? r1 + 1.f : __expf(r1));  // elu(x)+1
  q[base + 1] = (__bf16)(r2 > 0.f ? r2 + 1.f : __expf(r2));
}

// ---------------------------------------------------------------------------
// In-place sigmoid on a bf16 buffer.
// ---------------------------------------------------------------------------
__global__ __launch_bounds__(256) void sigmoid_inplace(__bf16* __restrict__ g,
                                                       int n) {
  int i = blockIdx.x * 256 + threadIdx.x;
  if (i < n) {
    float x = (float)g[i];
    g[i] = (__bf16)(1.f / (1.f + __expf(-x)));
  }
}

// ---------------------------------------------------------------------------
// Chunked linear attention scan. One block per (b,h); 8 waves; 64 chunks.
// State S (128x128 fp32) lives in WMMA accumulators: wave w owns e-columns
// [16w,16w+16). Per chunk:  S += K^T V ; Z += sum_t K ;
//                           num = Q S ; den = Q.Z + 1e-6 ; out = g*num/den.
// Fragment loads are batched ahead of each WMMA chain so multiple
// ds_load_b128 stay in flight instead of a wait per WMMA.
// ---------------------------------------------------------------------------
__global__ __launch_bounds__(256) void kla_attention(
    const __bf16* __restrict__ qb, const __bf16* __restrict__ kb,
    const __bf16* __restrict__ vb, const __bf16* __restrict__ gb,
    __bf16* __restrict__ ab) {
  const int bh = blockIdx.x;
  const int b  = bh >> 4;
  const int h  = bh & (HH - 1);

  // smem layout (49920 B): [0,32K) = Kt(16K)+Vt(16K) overlaid by Sb(32K);
  //                        [32K,48K) = Qs ; then Z (512B) + den (256B)
  __shared__ __align__(16) unsigned char smem[49920];
  __bf16* Kt = (__bf16*)smem;                 // [DK][CS] (d-major K^T)
  __bf16* Vt = (__bf16*)(smem + 16384);       // [DK][CS] (e-major V^T)
  __bf16* Sb = (__bf16*)smem;                 // [DK e][DK d], overlays Kt+Vt
  __bf16* Qs = (__bf16*)(smem + 32768);       // [CS][DK]
  float*  Zs = (float*)(smem + 49152);        // [DK]
  float*  dn = (float*)(smem + 49664);        // [CS]

  const int tid  = threadIdx.x;
  const int lane = tid & 31;
  const int w    = tid >> 5;   // e-strip [16w, 16w+16)
  const int half = lane >> 4;
  const int l16  = lane & 15;

  v8f S[8];
#pragma unroll
  for (int i = 0; i < 8; ++i) S[i] = {};
  if (tid < DKK) Zs[tid] = 0.f;
  __syncthreads();

  const size_t head_off = (size_t)h * DKK;
  const int NC = TT / CSZ;

  for (int c = 0; c < NC; ++c) {
    const size_t row0 = (size_t)b * TT + (size_t)c * CSZ;

    // ---- stage K^T, V^T (transposed) and Q (row-major) into LDS ----
#pragma unroll
    for (int it = 0; it < 4; ++it) {
      int v8i = tid + it * 256;        // 0..1023 eight-element chunks
      int t   = v8i >> 4;              // 0..63
      int d0  = (v8i & 15) * 8;        // 0..120
      const uint4 kvec = *reinterpret_cast<const uint4*>(kb + (row0 + t) * DD + head_off + d0);
      const uint4 vvec = *reinterpret_cast<const uint4*>(vb + (row0 + t) * DD + head_off + d0);
      const uint4 qvec = *reinterpret_cast<const uint4*>(qb + (row0 + t) * DD + head_off + d0);
      const __bf16* kp = (const __bf16*)&kvec;
      const __bf16* vp = (const __bf16*)&vvec;
#pragma unroll
      for (int j = 0; j < 8; ++j) {
        Kt[(d0 + j) * CSZ + t] = kp[j];
        Vt[(d0 + j) * CSZ + t] = vp[j];
      }
      *reinterpret_cast<uint4*>(Qs + t * DKK + d0) = qvec;
    }
    __syncthreads();

    // ---- S += K^T V  (8 d-tiles x 2 k-steps of 32 t) ; Z += sum_t K ----
#pragma unroll
    for (int ks = 0; ks < 2; ++ks) {
      Frag bfv;
      const __bf16* pb = &Vt[(w * 16 + l16) * CSZ + ks * 32 + half * 8];
      bfv.u4[0] = *reinterpret_cast<const uint4*>(pb);
      bfv.u4[1] = *reinterpret_cast<const uint4*>(pb + 16);
      Frag af[8];
#pragma unroll
      for (int mt = 0; mt < 8; ++mt) {
        const __bf16* pa = &Kt[(mt * 16 + l16) * CSZ + ks * 32 + half * 8];
        af[mt].u4[0] = *reinterpret_cast<const uint4*>(pa);
        af[mt].u4[1] = *reinterpret_cast<const uint4*>(pa + 16);
      }
#pragma unroll
      for (int mt = 0; mt < 8; ++mt)
        S[mt] = __builtin_amdgcn_wmma_f32_16x16x32_bf16(
            false, af[mt].v, false, bfv.v, (short)0, S[mt], false, false);
    }
    if (tid < DKK) {
      float z = 0.f;
      const __bf16* kr = &Kt[tid * CSZ];
#pragma unroll
      for (int t = 0; t < CSZ; ++t) z += (float)kr[t];
      Zs[tid] += z;
    }
    __syncthreads();

    // ---- spill S -> Sb (bf16, [e][d]) and compute den ----
#pragma unroll
    for (int mt = 0; mt < 8; ++mt) {
      int e = w * 16 + l16;
#pragma unroll
      for (int r = 0; r < 8; ++r) {
        int d = mt * 16 + r + half * 8;
        Sb[e * DKK + d] = (__bf16)S[mt][r];
      }
    }
    if (tid < CSZ) {
      float s = 0.f;
      const __bf16* qr = &Qs[tid * DKK];
#pragma unroll
      for (int d = 0; d < DKK; ++d) s += (float)qr[d] * Zs[d];
      dn[tid] = s + 1e-6f;
    }
    __syncthreads();

    // ---- num = Q S  (4 t-tiles x 4 k-steps of 32 d) ; gated output ----
    v8f num[4];
#pragma unroll
    for (int i = 0; i < 4; ++i) num[i] = {};
#pragma unroll
    for (int ks = 0; ks < 4; ++ks) {
      Frag bfv;
      const __bf16* pb = &Sb[(w * 16 + l16) * DKK + ks * 32 + half * 8];
      bfv.u4[0] = *reinterpret_cast<const uint4*>(pb);
      bfv.u4[1] = *reinterpret_cast<const uint4*>(pb + 16);
      Frag af[4];
#pragma unroll
      for (int tt = 0; tt < 4; ++tt) {
        const __bf16* pa = &Qs[(tt * 16 + l16) * DKK + ks * 32 + half * 8];
        af[tt].u4[0] = *reinterpret_cast<const uint4*>(pa);
        af[tt].u4[1] = *reinterpret_cast<const uint4*>(pa + 16);
      }
#pragma unroll
      for (int tt = 0; tt < 4; ++tt)
        num[tt] = __builtin_amdgcn_wmma_f32_16x16x32_bf16(
            false, af[tt].v, false, bfv.v, (short)0, num[tt], false, false);
    }
#pragma unroll
    for (int tt = 0; tt < 4; ++tt) {
      int e = w * 16 + l16;
#pragma unroll
      for (int r = 0; r < 8; ++r) {
        int tloc = tt * 16 + r + half * 8;
        float gv = (float)gb[(row0 + tloc) * DD + head_off + e];
        float o  = gv * num[tt][r] / dn[tloc];
        ab[(row0 + tloc) * DD + head_off + e] = (__bf16)o;
      }
    }
    __syncthreads();  // Sb/Qs reused next chunk
  }
}

// ---------------------------------------------------------------------------
// Host-side launch
// ---------------------------------------------------------------------------
extern "C" void kernel_launch(void* const* d_in, const int* in_sizes, int n_in,
                              void* d_out, int out_size, void* d_ws, size_t ws_size,
                              hipStream_t stream) {
  (void)in_sizes; (void)n_in; (void)out_size; (void)ws_size;
  const float* x  = (const float*)d_in[0];
  const float* Wq = (const float*)d_in[1];
  const float* bq = (const float*)d_in[2];
  const float* Wk = (const float*)d_in[3];
  const float* bk = (const float*)d_in[4];
  const float* Wv = (const float*)d_in[5];
  const float* bv = (const float*)d_in[6];
  const float* Wg = (const float*)d_in[7];
  const float* bg = (const float*)d_in[8];
  const float* Wo = (const float*)d_in[9];
  const float* bo = (const float*)d_in[10];

  char* ws = (char*)d_ws;
  const size_t MB = 1u << 20;
  __bf16* xb  = (__bf16*)(ws);                // 64 MB  x bf16
  __bf16* wqb = (__bf16*)(ws + 64 * MB);      // 8 MB each
  __bf16* wkb = (__bf16*)(ws + 72 * MB);
  __bf16* wvb = (__bf16*)(ws + 80 * MB);
  __bf16* wgb = (__bf16*)(ws + 88 * MB);
  __bf16* wob = (__bf16*)(ws + 96 * MB);
  __bf16* qbuf = (__bf16*)(ws + 104 * MB);    // 64 MB each
  __bf16* kbuf = (__bf16*)(ws + 168 * MB);
  __bf16* vbuf = (__bf16*)(ws + 232 * MB);
  __bf16* gbuf = (__bf16*)(ws + 296 * MB);
  __bf16* abuf = (__bf16*)(ws + 360 * MB);    // total 424 MB

  const int nX = MM * DD;   // 33554432
  const int nW = DD * DD;   // 4194304

  f32_to_bf16_kernel<<<nX / 256, 256, 0, stream>>>(x, xb, nX);
  f32_to_bf16_kernel<<<nW / 256, 256, 0, stream>>>(Wq, wqb, nW);
  f32_to_bf16_kernel<<<nW / 256, 256, 0, stream>>>(Wk, wkb, nW);
  f32_to_bf16_kernel<<<nW / 256, 256, 0, stream>>>(Wv, wvb, nW);
  f32_to_bf16_kernel<<<nW / 256, 256, 0, stream>>>(Wg, wgb, nW);
  f32_to_bf16_kernel<<<nW / 256, 256, 0, stream>>>(Wo, wob, nW);

  dim3 gg(DD / 128, MM / 128);  // (16, 128)
  gemm_bf16_wmma<true><<<gg, 256, 0, stream>>>(xb, wqb, bq, qbuf, MM, DD, DD);
  gemm_bf16_wmma<true><<<gg, 256, 0, stream>>>(xb, wkb, bk, kbuf, MM, DD, DD);
  gemm_bf16_wmma<true><<<gg, 256, 0, stream>>>(xb, wvb, bv, vbuf, MM, DD, DD);
  gemm_bf16_wmma<true><<<gg, 256, 0, stream>>>(xb, wgb, bg, gbuf, MM, DD, DD);

  const int npairs = MM * DD / 2;  // 16777216
  rope_elu_inplace<<<npairs / 256, 256, 0, stream>>>(qbuf, npairs);
  rope_elu_inplace<<<npairs / 256, 256, 0, stream>>>(kbuf, npairs);
  sigmoid_inplace<<<nX / 256, 256, 0, stream>>>(gbuf, nX);

  kla_attention<<<BB * HH, 256, 0, stream>>>(qbuf, kbuf, vbuf, gbuf, abuf);

  gemm_bf16_wmma<false><<<gg, 256, 0, stream>>>(abuf, wob, bo, d_out, MM, DD, DD);
}
